// REGRESSION_WIDE_2405181686069
// MI455X (gfx1250) — matrix operand order repaired and checked by reading, not compiled, not run on hardware
//
#include <hip/hip_runtime.h>
#include <hip/hip_bf16.h>

// ---- problem constants ----
#define F_DIM   88
#define WIN     64
#define KTOT    (WIN * F_DIM)        // 5632
#define N_BATCH 128
#define T_LEN   2048
#define TM      128                  // t-rows per block
#define KC      128                  // K-chunk staged in LDS
#define NKC     (KTOT / KC)          // 44 chunks
#define NT      6                    // 96 output cols = 6 tiles of 16 (88 valid)
#define OPAD    96
#define WROW    136                  // padded LDS row stride (f16), 272B row pitch
#define XS_ELEMS ((TM + WIN - 1) * F_DIM)   // 191*88 = 16808 f16
#define XS_CHUNKS (XS_ELEMS / 8)            // 2101 16-byte chunks
#define WCH_CHUNKS (OPAD * (KC / 8))        // 96*16 = 1536 16-byte chunks

typedef _Float16 hv8  __attribute__((ext_vector_type(8)));
typedef _Float16 hv16 __attribute__((ext_vector_type(16)));
typedef _Float16 hv4  __attribute__((ext_vector_type(4)));
typedef float    fv8  __attribute__((ext_vector_type(8)));
typedef float    fv4  __attribute__((ext_vector_type(4)));

// vector-of-4-int pointee type the async builtin expects
typedef int iv4 __attribute__((vector_size(16)));
typedef __attribute__((address_space(1))) iv4 giv4;   // global
typedef __attribute__((address_space(3))) iv4 liv4;   // LDS

// ---- async 16B global->LDS copy (CDNA5 ASYNCcnt path), sync fallback ----
__device__ __forceinline__ void cp16_g2l(const _Float16* __restrict__ g,
                                         _Float16* __restrict__ l) {
#if __has_builtin(__builtin_amdgcn_global_load_async_to_lds_b128)
  __builtin_amdgcn_global_load_async_to_lds_b128(
      (giv4*)(iv4*)g, (liv4*)(iv4*)l, /*offset=*/0, /*cpol=*/0);
#else
  *(hv8*)l = *(const hv8*)g;
#endif
}

__device__ __forceinline__ void wait_async_all() {
#if __has_builtin(__builtin_amdgcn_s_wait_asynccnt)
  __builtin_amdgcn_s_wait_asynccnt(0);
#else
  asm volatile("s_wait_asynccnt 0" ::: "memory");
#endif
}

// ---- prep: W (88 x 5632) f32 -> f16 padded to 96 rows (zeros) ----
__global__ __launch_bounds__(256) void prep_w(const float* __restrict__ W,
                                              _Float16* __restrict__ wh) {
  int i = blockIdx.x * 256 + threadIdx.x;            // < 96*5632 = 540672
  int o = i / KTOT;
  int k = i - o * KTOT;
  float v = (o < F_DIM) ? W[o * KTOT + k] : 0.0f;
  wh[i] = (_Float16)v;
}

// ---- prep: x f32 -> f16, 4 elements per thread ----
__global__ __launch_bounds__(256) void prep_x(const float* __restrict__ x,
                                              _Float16* __restrict__ xh) {
  size_t i = (size_t)blockIdx.x * 256 + threadIdx.x; // < 23068672/4
  fv4 v = *(const fv4*)(x + 4 * i);
  hv4 h = __builtin_convertvector(v, hv4);
  *(hv4*)(xh + 4 * i) = h;
}

// ---- main: causal window GEMM via V_WMMA_F32_16X16X32_F16,
//      W chunks double-buffered through LDS on the ASYNCcnt path ----
__global__ __launch_bounds__(256) void conv_wmma(
    const _Float16* __restrict__ xh,   // (N, T, F) f16
    const _Float16* __restrict__ wh,   // (96, 5632) f16, rows 88..95 zero
    const float*    __restrict__ bias, // (88,)
    float* __restrict__ out0,
    float* __restrict__ out1) {
  __shared__ _Float16 xs[XS_ELEMS];           // frames t0-64 .. t0+126
  __shared__ _Float16 wsh[2][OPAD * WROW];    // ping-pong W chunk

  const int tid  = threadIdx.x;
  const int wave = tid >> 5;
  const int lane = tid & 31;
  const int n    = blockIdx.y;
  const int t0   = blockIdx.x * TM;

  // ---------- stage the x window (once) + W chunk 0 ----------
  // xs[e] <-> xh[gbase + e]; zeros for the left causal pad of the first tile.
  const long gbase = ((long)n * T_LEN + (long)(t0 - WIN)) * F_DIM;
  const int zeroChunks = (t0 == 0) ? (WIN * F_DIM) / 8 : 0;   // 704 or 0
  for (int c = tid; c < XS_CHUNKS; c += 256) {
    int e = c * 8;
    if (c < zeroChunks) {
      hv8 z = {};
      *(hv8*)&xs[e] = z;
    } else {
      cp16_g2l(xh + gbase + e, &xs[e]);
    }
  }
  for (int c = tid; c < WCH_CHUNKS; c += 256) {
    int o   = c >> 4;
    int cir = (c & 15) * 8;
    cp16_g2l(wh + (size_t)o * KTOT + cir, &wsh[0][o * WROW + cir]);
  }
  wait_async_all();
  __syncthreads();

  // ---------- fragment addressing (wave32 WMMA layouts) ----------
  // A (16x32 f16): lane m = lane&15; runs at kbase, kbase+16, kbase = (lane>=16)?8:0
  // B (32x16 f16): lane col = lane&15; runs at kb, kb+16, kb = (lane>=16)?8:0
  const int m_loc = wave * 16 + (lane & 15);
  const int kA    = (lane >> 4) * 8;
  const int kB    = (lane >> 4) * 8;
  const int oB    = (lane & 15);

  fv8 acc[NT];
#pragma unroll
  for (int i = 0; i < NT; ++i) acc[i] = (fv8){};

  for (int kc = 0; kc < NKC; ++kc) {               // 44 chunks
    const int cur = kc & 1;
    // issue async stage of the NEXT chunk into the other buffer,
    // overlapping the DMA with this chunk's 24 WMMAs
    if (kc + 1 < NKC) {
      const int k0n = (kc + 1) * KC;
      _Float16* dst = &wsh[cur ^ 1][0];
      for (int c = tid; c < WCH_CHUNKS; c += 256) {
        int o   = c >> 4;
        int cir = (c & 15) * 8;
        cp16_g2l(wh + (size_t)o * KTOT + k0n + cir, dst + o * WROW + cir);
      }
    }

#pragma unroll
    for (int ks = 0; ks < KC / 32; ++ks) {         // 4 k-steps of 32
      const int kg = kc * KC + ks * 32;
      union { hv16 v; hv8 h[2]; } A;
      const _Float16* ap = &xs[m_loc * F_DIM + kg + kA];
      A.h[0] = *(const hv8*)ap;                    // K = kbase .. kbase+7
      A.h[1] = *(const hv8*)(ap + 16);             // K = kbase+16 .. kbase+23
#pragma unroll
      for (int nt = 0; nt < NT; ++nt) {
        union { hv16 v; hv8 h[2]; } B;
        const _Float16* bp = &wsh[cur][(nt * 16 + oB) * WROW + ks * 32 + kB];
        B.h[0] = *(const hv8*)bp;                  // K = kb .. kb+7
        B.h[1] = *(const hv8*)(bp + 16);           // K = kb+16 .. kb+23
        acc[nt] = __builtin_amdgcn_wmma_f32_16x16x32_f16(
            false, A.v, false, B.v, (short)0, acc[nt], false, false);
      }
    }

    wait_async_all();        // next chunk landed
    __syncthreads();         // all waves done reading cur before it is reused
  }

  // ---------- epilogue: +bias, store both tuple copies ----------
  // C/D layout: VGPR r -> M = (lane<16 ? r : r+8), N = lane&15
#pragma unroll
  for (int nt = 0; nt < NT; ++nt) {
    const int o = nt * 16 + (lane & 15);
    if (o < F_DIM) {
      const float bb = bias[o];
#pragma unroll
      for (int r = 0; r < 8; ++r) {
        const int mm = (lane < 16) ? r : (r + 8);
        const size_t idx =
            ((size_t)n * T_LEN + (size_t)(t0 + wave * 16 + mm)) * F_DIM + o;
        const float v = acc[nt][r] + bb;
        out0[idx] = v;
        out1[idx] = v;
      }
    }
  }
}

extern "C" void kernel_launch(void* const* d_in, const int* in_sizes, int n_in,
                              void* d_out, int out_size, void* d_ws, size_t ws_size,
                              hipStream_t stream) {
  (void)in_sizes; (void)n_in; (void)out_size; (void)ws_size;
  const float* x = (const float*)d_in[0];   // (128, 2048, 88)
  const float* W = (const float*)d_in[1];   // (88, 5632)
  const float* b = (const float*)d_in[2];   // (88,)
  float* out = (float*)d_out;

  const size_t WH_ELEMS = (size_t)OPAD * KTOT;                 // 540672
  const size_t X_ELEMS  = (size_t)N_BATCH * T_LEN * F_DIM;     // 23068672
  _Float16* wh = (_Float16*)d_ws;
  _Float16* xh = (_Float16*)((char*)d_ws + WH_ELEMS * sizeof(_Float16));

  prep_w<<<(int)(WH_ELEMS / 256), 256, 0, stream>>>(W, wh);
  prep_x<<<(int)(X_ELEMS / 4 / 256), 256, 0, stream>>>(x, xh);

  dim3 grid(T_LEN / TM, N_BATCH);   // (16, 128) = 2048 blocks, 8 waves each
  conv_wmma<<<grid, 256, 0, stream>>>(xh, wh, b, out, out + X_ELEMS);
}
